// QuantileDilatedDenseGINConv_45818711113827
// MI455X (gfx1250) — matrix operand order repaired and checked by reading, not compiled, not run on hardware
//
#include <hip/hip_runtime.h>
#include <hip/hip_bf16.h>

typedef __attribute__((ext_vector_type(16))) _Float16 v16h;
typedef __attribute__((ext_vector_type(8)))  float    v8f;
typedef unsigned __attribute__((ext_vector_type(4)))  v4u;
typedef unsigned __attribute__((ext_vector_type(8)))  v8u;

#define BB 8
#define NN 512
#define CC 64
#define HH 128
#define TT 10
#define KK 2
#define WORDS 16          // NN / 32
#define NVSTAGE 160       // LDS-staged neighbor rows (typ. nv ~ 60)

static __device__ __forceinline__ unsigned ballot32(bool p) {
  return __builtin_amdgcn_ballot_w32(p);
}

// Issue a 2D TDM load: tensor (dim1 rows x dim0 cols) of 4-byte elements,
// tile == whole tensor, contiguous (stride0 = dim0), into LDS at lds_addr.
// D# group0/group1 packed per CDNA5 ISA 8.3/8.4; VADDR2/3 omitted (<=2D).
static __device__ __forceinline__ void tdm_load_2d_f32(
    unsigned lds_addr, const void* gptr, unsigned dim0, unsigned dim1) {
  unsigned long long ga = (unsigned long long)gptr;
  v4u g0;
  g0[0] = 1u;                                  // count=1, user mode
  g0[1] = lds_addr;                            // LDS byte address
  g0[2] = (unsigned)ga;                        // global_addr[31:0]
  g0[3] = (unsigned)(ga >> 32) | (2u << 30);   // global_addr[56:32] | type=2
  v8u g1;
  g1[0] = (2u << 16);                          // data_size=2 (4 bytes)
  g1[1] = (dim0 & 0xFFFFu) << 16;              // tensor_dim0[15:0]
  g1[2] = (dim0 >> 16) | ((dim1 & 0xFFFFu) << 16); // dim0[31:16] | dim1[15:0]
  g1[3] = (dim1 >> 16) | (dim0 << 16);         // dim1[31:16] | tile_dim0
  g1[4] = dim1;                                // tile_dim1 (tile_dim2=0)
  g1[5] = dim0;                                // tensor_dim0_stride[31:0]
  g1[6] = 0u;                                  // stride0[47:32], stride1[15:0]
  g1[7] = 0u;
  asm volatile("tensor_load_to_lds %0, %1" :: "s"(g0), "s"(g1) : "memory");
}

// ---------------------------------------------------------------------------
// Kernel 1: per-row neighbor bitmask + dilated-removal bitmask (pre-symmetry)
// One wave per (b,i) row; ballot -> 32-bit mask words, rank via prefix popc.
// ---------------------------------------------------------------------------
__global__ __launch_bounds__(256) void k_rem(
    const float* __restrict__ adj, const unsigned char* __restrict__ mask,
    unsigned* __restrict__ neighW, unsigned* __restrict__ remW) {
  int wid  = (int)((blockIdx.x * blockDim.x + threadIdx.x) >> 5);
  int lane = (int)(threadIdx.x & 31);
  if (wid >= BB * NN) return;
  int i = wid & (NN - 1);
  const float* arow = adj + (size_t)wid * NN;

  unsigned nw[WORDS];
  int nb = 0;
#pragma unroll
  for (int w = 0; w < WORDS; ++w) {
    int j = w * 32 + lane;
    __builtin_prefetch(arow + j + 64, 0, 0);
    float a  = arow[j];
    bool  ne = (j != i) && (a > 0.0f);
    unsigned m = ballot32(ne);
    nw[w] = m;
    nb += __popc(m);
  }
  int skip = (nb > TT) ? (nb + KK - 1) / KK : 1;
  if (skip < 1) skip = 1;
  bool mk = mask[wid] != 0;

  unsigned lmask = (1u << lane) - 1u;
  int pref = 0;
#pragma unroll
  for (int w = 0; w < WORDS; ++w) {
    unsigned m   = nw[w];
    int     rank = pref + __popc(m & lmask);
    bool    ne   = (m >> lane) & 1u;
    bool    rm   = ne && (skip > 1) && mk && ((rank % skip) == skip - 1);
    unsigned r   = ballot32(rm);
    if (lane == 0) {
      neighW[(size_t)wid * WORDS + w] = m;
      remW  [(size_t)wid * WORDS + w] = r;
    }
    pref += __popc(m);
  }
}

// ---------------------------------------------------------------------------
// Kernel 2: symmetrize rem (rem | rem^T), build nd = (neigh|diag) & ~rem_sym
// ---------------------------------------------------------------------------
__global__ __launch_bounds__(256) void k_sym(
    const unsigned* __restrict__ neighW, const unsigned* __restrict__ remW,
    unsigned* __restrict__ ndW, int* __restrict__ nvOut) {
  int wid  = (int)((blockIdx.x * blockDim.x + threadIdx.x) >> 5);
  int lane = (int)(threadIdx.x & 31);
  if (wid >= BB * NN) return;
  int b = wid >> 9;
  int i = wid & (NN - 1);
  int      iw = i >> 5;
  unsigned ib = (unsigned)(i & 31);

  int nv = 0;
#pragma unroll
  for (int w = 0; w < WORDS; ++w) {
    int j = w * 32 + lane;
    unsigned tb   = (remW[(size_t)(b * NN + j) * WORDS + iw] >> ib) & 1u;
    unsigned remT = ballot32(tb != 0u);
    unsigned m    = neighW[(size_t)wid * WORDS + w];
    unsigned r    = remW  [(size_t)wid * WORDS + w];
    unsigned diag = (w == iw) ? (1u << ib) : 0u;
    unsigned nd   = (m | diag) & ~(r | remT);
    if (lane == 0) ndW[(size_t)wid * WORDS + w] = nd;
    nv += __popc(nd);
  }
  if (lane == 0) nvOut[wid] = nv;
}

// ---------------------------------------------------------------------------
// Kernel 3: per-node weighted-quantile aggregation via rank-count selection.
// Neighbor rows gathered into LDS with GLOBAL_LOAD_ASYNC_TO_LDS_B128
// (per-lane async gather, ASYNCcnt), then O(nv^2) rank counting for the
// 6 needed order statistics (no sort).
// ---------------------------------------------------------------------------
__global__ __launch_bounds__(256) void k_quant(
    const float* __restrict__ x, const unsigned* __restrict__ ndW,
    float* __restrict__ outws) {
  __shared__ int   idxS[NN];
  __shared__ __align__(16) float vals[NVSTAGE * CC];
  __shared__ float res[6 * CC];
  __shared__ int   pref[WORDS + 1];
  __shared__ int   targS[6];
  __shared__ float fracS[3];

  int row = (int)blockIdx.x;
  int b   = row >> 9;
  int t   = (int)threadIdx.x;

  if (t < WORDS) pref[t] = __popc(ndW[(size_t)row * WORDS + t]);
  __syncthreads();
  if (t == 0) {
    int run = 0;
    for (int w = 0; w < WORDS; ++w) { int c = pref[w]; pref[w] = run; run += c; }
    pref[WORDS] = run;
    int nv = run;
    const float taus[3] = {0.25f, 0.5f, 0.75f};
    int nvm1 = nv - 1 > 0 ? nv - 1 : 0;
    for (int q = 0; q < 3; ++q) {
      float pos = taus[q] * (float)nvm1;
      int lo = (int)floorf(pos);
      int hi = (int)ceilf(pos);
      targS[2 * q] = lo; targS[2 * q + 1] = hi;
      fracS[q] = pos - (float)lo;
    }
  }
  __syncthreads();
  if (t < WORDS) {
    unsigned m = ndW[(size_t)row * WORDS + t];
    int base = pref[t];
    while (m) {
      int bp = __ffs(m) - 1;
      idxS[base++] = t * 32 + bp;
      m &= (m - 1u);
    }
  }
  __syncthreads();

  int nv   = pref[WORDS];
  int c    = t & 63;
  int part = t >> 6;                      // 4 candidate partitions per channel
  int stage = nv < NVSTAGE ? nv : NVSTAGE;

  // Async gather: row k is 256B = 16 chunks of 16B; thread t>>4 walks rows,
  // t&15 picks the 16B chunk. Straight global->LDS, no VGPR round-trip.
  {
    int chunk = t & 15;
    for (int k = t >> 4; k < stage; k += 16) {
      unsigned goff = (unsigned)((b * NN + idxS[k]) * (CC * 4) + (chunk << 4));
      unsigned ldsa = (unsigned)(unsigned long long)(const void*)
                          &vals[(size_t)k * CC + (chunk << 2)];
      asm volatile("global_load_async_to_lds_b128 %0, %1, %2"
                   :: "v"(ldsa), "v"(goff), "s"(x) : "memory");
    }
    asm volatile("s_wait_asynccnt 0x0" ::: "memory");
  }
  __syncthreads();

  for (int k = part; k < nv; k += 4) {
    float v = (k < NVSTAGE) ? vals[k * CC + c]
                            : x[(size_t)(b * NN + idxS[k]) * CC + c];
    int less = 0, eq = 0;
    for (int m = 0; m < nv; ++m) {
      float u = (m < NVSTAGE) ? vals[m * CC + c]
                              : x[(size_t)(b * NN + idxS[m]) * CC + c];
      less += (u < v);
      eq   += (u == v);
    }
#pragma unroll
    for (int q = 0; q < 6; ++q) {
      int r = targS[q];
      if (less <= r && r < less + eq) res[q * CC + c] = v;  // benign tie race
    }
  }
  __syncthreads();

  if (t < CC) {
    const float wts[3] = {0.25f, 0.5f, 0.25f};
    float agg = 0.0f;
#pragma unroll
    for (int q = 0; q < 3; ++q) {
      float vlo = res[(2 * q) * CC + t];
      float vhi = res[(2 * q + 1) * CC + t];
      agg += wts[q] * (vlo + fracS[q] * (vhi - vlo));
    }
    float xv = x[(size_t)row * CC + t];
    outws[(size_t)row * CC + t] = xv + ((nv > 0) ? agg : 0.0f);
  }
}

// ---------------------------------------------------------------------------
// Kernel 4: fused GIN MLP  y = relu(out@W1 + b1)@W2 + b2, masked.
// Weights DMA'd into LDS by the Tensor Data Mover (one tensor_load_to_lds per
// phase, TENSORcnt-tracked), WMMA fragments built from LDS.
// Block = 16-row stripe, 8 waves. GEMM1: each wave one 16x16 h-tile
// (2x v_wmma_f32_16x16x32_f16, K=64), relu+bias -> f16 LDS. GEMM2: waves 0-3,
// 4 WMMAs each (K=128), bias+mask on store.
// ---------------------------------------------------------------------------
__global__ __launch_bounds__(256) void k_mlp(
    const float* __restrict__ outws,
    const float* __restrict__ W1, const float* __restrict__ b1,
    const float* __restrict__ W2, const float* __restrict__ b2,
    const unsigned char* __restrict__ mask, float* __restrict__ y) {
  __shared__ __align__(16) float w_lds[CC * HH];   // 32KB: W1, then W2
  __shared__ _Float16 hld[16 * HH];                // 4KB

  int t    = (int)threadIdx.x;
  int wave = t >> 5, lane = t & 31;
  int lm   = lane & 15, hiH = lane >> 4;
  int bm   = (int)blockIdx.x * 16;
  unsigned wbase = (unsigned)(unsigned long long)(const void*)&w_lds[0];

  if (wave == 0) {                     // TDM: W1 (64 rows x 128 cols) -> LDS
    tdm_load_2d_f32(wbase, W1, HH, CC);
    __builtin_amdgcn_s_wait_tensorcnt(0);
  }
  __syncthreads();

  { // GEMM1: h[:, wave*16 .. wave*16+15]
    int col = wave * 16 + lm;
    v8f acc = {};
#pragma unroll
    for (int ks = 0; ks < CC; ks += 32) {
      v16h a, bf;
#pragma unroll
      for (int e = 0; e < 16; ++e) {
        int ka = ks + hiH * 8 + e + (e >= 8 ? 8 : 0);
        a[e] = (_Float16)outws[(size_t)(bm + lm) * CC + ka];
        int kb = ks + hiH * 8 + e + (e >= 8 ? 8 : 0);
        bf[e] = (_Float16)w_lds[kb * HH + col];
      }
      acc = __builtin_amdgcn_wmma_f32_16x16x32_f16(false, a, false, bf,
                                                   (short)0, acc, false, false);
    }
    float bias = b1[col];
#pragma unroll
    for (int r = 0; r < 8; ++r) {
      int M = r + hiH * 8;
      float hv = acc[r] + bias;
      hld[M * HH + col] = (_Float16)(hv > 0.0f ? hv : 0.0f);
    }
  }
  __syncthreads();

  if (wave == 0) {                     // TDM: W2 (128 rows x 64 cols) -> LDS
    tdm_load_2d_f32(wbase, W2, CC, HH);
    __builtin_amdgcn_s_wait_tensorcnt(0);
  }
  __syncthreads();

  if (wave < 4) { // GEMM2: y[:, wave*16 .. wave*16+15]
    int col = wave * 16 + lm;
    v8f acc = {};
#pragma unroll
    for (int ks = 0; ks < HH; ks += 32) {
      v16h a, bf;
#pragma unroll
      for (int e = 0; e < 16; ++e) {
        int ka = ks + hiH * 8 + e + (e >= 8 ? 8 : 0);
        a[e] = hld[lm * HH + ka];
        int kb = ks + hiH * 8 + e + (e >= 8 ? 8 : 0);
        bf[e] = (_Float16)w_lds[kb * CC + col];
      }
      acc = __builtin_amdgcn_wmma_f32_16x16x32_f16(false, a, false, bf,
                                                   (short)0, acc, false, false);
    }
    float bias = b2[col];
#pragma unroll
    for (int r = 0; r < 8; ++r) {
      int M   = r + hiH * 8;
      int row = bm + M;
      float yv = acc[r] + bias;
      yv = mask[row] ? yv : 0.0f;
      y[(size_t)row * CC + col] = yv;
    }
  }
}

// ---------------------------------------------------------------------------
extern "C" void kernel_launch(void* const* d_in, const int* in_sizes, int n_in,
                              void* d_out, int out_size, void* d_ws, size_t ws_size,
                              hipStream_t stream) {
  (void)in_sizes; (void)n_in; (void)out_size; (void)ws_size;
  const float*         x    = (const float*)d_in[0];
  const float*         adj  = (const float*)d_in[1];
  const unsigned char* mask = (const unsigned char*)d_in[2]; // numpy bool: 1B
  const float*         W1   = (const float*)d_in[3];
  const float*         b1   = (const float*)d_in[4];
  const float*         W2   = (const float*)d_in[5];
  const float*         b2   = (const float*)d_in[6];
  float*               y    = (float*)d_out;

  char* ws = (char*)d_ws;
  const size_t ROWS = (size_t)BB * NN;                     // 4096
  unsigned* neighW = (unsigned*)(ws);                      // 256 KB
  unsigned* remW   = (unsigned*)(ws + ROWS * WORDS * 4);   // 256 KB
  unsigned* ndW    = (unsigned*)(ws + 2 * ROWS * WORDS * 4);
  int*      nvOut  = (int*)     (ws + 3 * ROWS * WORDS * 4);
  float*    outws  = (float*)   (ws + 3 * ROWS * WORDS * 4 + ROWS * 4);

  k_rem  <<<dim3((unsigned)(ROWS / 8)), dim3(256), 0, stream>>>(adj, mask, neighW, remW);
  k_sym  <<<dim3((unsigned)(ROWS / 8)), dim3(256), 0, stream>>>(neighW, remW, ndW, nvOut);
  k_quant<<<dim3((unsigned)ROWS),       dim3(256), 0, stream>>>(x, ndW, outws);
  k_mlp  <<<dim3((unsigned)(ROWS / 16)),dim3(256), 0, stream>>>(outws, W1, b1, W2, b2, mask, y);
}
